// pooled_moe_22067541967821
// MI455X (gfx1250) — matrix operand order repaired and checked by reading, not compiled, not run on hardware
//
#include <hip/hip_runtime.h>

// ---- problem constants (B=4, L=576, N=3, D=1024, H=4096, E=3) ----
#define BB 4
#define LL 576
#define NN 3
#define DD 1024
#define HH 4096
#define EE 3
#define SS (BB*LL*NN)        // 6912 tokens
#define TPB (NN*LL)          // 1728 tokens per batch
#define CAP (SS/EE)          // 2304 capacity
#define KB  (DD/32)          // 32 k-blocks of 32

typedef __attribute__((ext_vector_type(16))) __bf16 v16bf;
typedef __attribute__((ext_vector_type(8)))  __bf16 v8bf;
typedef __attribute__((ext_vector_type(8)))  float  v8f;

// ------------------------------------------------------------------
// Zero-init (for padded bf16 A matrices; ws is poisoned each run)
// ------------------------------------------------------------------
__global__ void k_zero(unsigned int* __restrict__ p, int n) {
    int i = blockIdx.x * 256 + threadIdx.x;
    if (i < n) p[i] = 0u;
}

// ------------------------------------------------------------------
// Pack f32 weight [K x ncols] into bf16 WMMA-B fragment layout:
// dst[((tile*kblocks + kb)*32 + lane)*16 + j], lane = n + (k>=16?16:0),
// j = k&15 (contiguous k order, matches load_afrag). One block per 32x16 tile.
// ------------------------------------------------------------------
__global__ void k_pack(const float* __restrict__ src, __bf16* __restrict__ dst,
                       int ncols, int kblocks)
{
    int blk  = blockIdx.x;              // = tile*kblocks + kb
    int tile = blk / kblocks;
    int kb   = blk % kblocks;
    int t    = threadIdx.x;
    #pragma unroll
    for (int h = 0; h < 2; h++) {
        int idx = t + h*256;            // 0..511
        int k   = idx >> 4;             // 0..31
        int n   = idx & 15;
        float v = src[(size_t)(kb*32 + k)*ncols + tile*16 + n];
        int lane = n + ((k & 16) ? 16 : 0);
        int j    = k & 15;
        dst[((size_t)blk*32 + lane)*16 + j] = (__bf16)v;
    }
}

// ------------------------------------------------------------------
// Kernel 1: gating logits + softmax + argmax. One wave per token.
// ------------------------------------------------------------------
__global__ void k_gating(const float* __restrict__ feat,
                         const float* __restrict__ wg,
                         float* __restrict__ gates,
                         float* __restrict__ gate_top,
                         int*   __restrict__ eidx)
{
    int tid  = threadIdx.x;
    int lane = tid & 31;
    int s    = blockIdx.x * 8 + (tid >> 5);
    if (s >= SS) return;
    int b = s / TPB;
    int r = s % TPB;
    int n = r / LL;
    int l = r % LL;
    const float* xrow = feat + (size_t)((b*LL + l)*NN + n) * DD;
    float a0 = 0.f, a1 = 0.f, a2 = 0.f;
    for (int d = lane; d < DD; d += 32) {
        float xv = xrow[d];
        a0 += xv * wg[d*EE + 0];
        a1 += xv * wg[d*EE + 1];
        a2 += xv * wg[d*EE + 2];
    }
    for (int off = 16; off >= 1; off >>= 1) {
        a0 += __shfl_xor(a0, off, 32);
        a1 += __shfl_xor(a1, off, 32);
        a2 += __shfl_xor(a2, off, 32);
    }
    if (lane == 0) {
        float m  = fmaxf(a0, fmaxf(a1, a2));
        float e0 = expf(a0 - m), e1 = expf(a1 - m), e2 = expf(a2 - m);
        float inv = 1.0f / (e0 + e1 + e2);
        float g0 = e0*inv, g1 = e1*inv, g2 = e2*inv;
        gates[s*3+0] = g0; gates[s*3+1] = g1; gates[s*3+2] = g2;
        int best = 0; float bg = g0;
        if (g1 > bg) { bg = g1; best = 1; }
        if (g2 > bg) { bg = g2; best = 2; }
        gate_top[s] = bg;
        eidx[s]     = best;
    }
}

// ------------------------------------------------------------------
// Kernel 2: single-block capacity scan (exact reference cumsum),
// l_aux, per-(b,e) gate sums, per-batch counts. Deterministic.
// ------------------------------------------------------------------
__global__ void k_scan(const float* __restrict__ gates,
                       const float* __restrict__ gate_top,
                       const int*   __restrict__ eidx,
                       const unsigned char* __restrict__ mask,
                       float* __restrict__ wgate,
                       float* __restrict__ gsum,   // [B*E]
                       float* __restrict__ cntb,   // [B]
                       float* __restrict__ laux_out)
{
    __shared__ int   sc[EE][256];
    __shared__ float sg[256*EE];
    __shared__ float sm[256*EE];
    __shared__ int   smk[256];
    int t  = threadIdx.x;
    int s0 = t * 27;

    int c[EE] = {0,0,0};
    for (int i = 0; i < 27; i++) c[eidx[s0+i]]++;
    sc[0][t] = c[0]; sc[1][t] = c[1]; sc[2][t] = c[2];
    __syncthreads();
    for (int off = 1; off < 256; off <<= 1) {
        int t0 = (t >= off) ? sc[0][t-off] : 0;
        int t1 = (t >= off) ? sc[1][t-off] : 0;
        int t2 = (t >= off) ? sc[2][t-off] : 0;
        __syncthreads();
        sc[0][t] += t0; sc[1][t] += t1; sc[2][t] += t2;
        __syncthreads();
    }
    int base[EE] = { sc[0][t]-c[0], sc[1][t]-c[1], sc[2][t]-c[2] };
    int tot[EE]  = { sc[0][255],    sc[1][255],    sc[2][255]    };

    float gl[EE] = {0.f,0.f,0.f};
    for (int i = 0; i < 27; i++) {
        int s = s0 + i;
        int e = eidx[s];
        int loc = base[e]++;
        float w = (loc < CAP) ? gate_top[s] : 0.0f;
        wgate[s] = w;
        gl[e] += w;
    }
    sg[t*3+0] = gl[0]; sg[t*3+1] = gl[1]; sg[t*3+2] = gl[2];

    float gm0 = 0.f, gm1 = 0.f, gm2 = 0.f;
    for (int i = 0; i < 27; i++) {
        int s = s0 + i;
        gm0 += gates[s*3+0]; gm1 += gates[s*3+1]; gm2 += gates[s*3+2];
    }
    sm[t*3+0] = gm0; sm[t*3+1] = gm1; sm[t*3+2] = gm2;

    int mc = 0;
    for (int i = 0; i < 27; i++) mc += (mask[s0+i] == 0) ? 1 : 0;
    smk[t] = mc;
    __syncthreads();

    if (t < BB*EE) {
        int b = t / EE, e = t % EE;
        float a = 0.f;
        for (int j = 0; j < 64; j++) a += sg[(b*64 + j)*3 + e];
        gsum[b*EE + e] = a;
    }
    if (t < BB) {
        int a = 0;
        for (int j = 0; j < 64; j++) a += smk[t*64 + j];
        cntb[t] = (float)(a < 1 ? 1 : a);
    }
    if (t == 0) {
        float me0 = 0.f, me1 = 0.f, me2 = 0.f;
        for (int j = 0; j < 256; j++) { me0 += sm[j*3]; me1 += sm[j*3+1]; me2 += sm[j*3+2]; }
        float invS = 1.0f / (float)SS;
        *laux_out = (float)EE * (me0*invS*((float)tot[0]*invS) +
                                 me1*invS*((float)tot[1]*invS) +
                                 me2*invS*((float)tot[2]*invS));
    }
}

// ------------------------------------------------------------------
// Kernel 3: xbarA[e][b][d] = sum over kept tokens of gate * x (bf16,
// rows 4..15 pre-zeroed). One pass over features, coalesced in d.
// ------------------------------------------------------------------
__global__ void k_xbar(const float* __restrict__ feat,
                       const float* __restrict__ wgate,
                       const int*   __restrict__ eidx,
                       __bf16* __restrict__ xbarA)   // [E][16][D]
{
    __shared__ float shw[TPB];
    __shared__ int   shi[TPB];
    int blk = blockIdx.x;
    int b     = blk >> 2;
    int dbase = (blk & 3) * 256;
    int t = threadIdx.x;
    for (int i = t; i < TPB; i += 256) {
        shw[i] = wgate[b*TPB + i];
        shi[i] = eidx [b*TPB + i];
    }
    __syncthreads();
    int d = dbase + t;
    float a0 = 0.f, a1 = 0.f, a2 = 0.f;
    for (int tok = 0; tok < TPB; tok++) {
        float w = shw[tok];
        if (w != 0.0f) {
            int n = tok / LL, l = tok % LL;
            float x = feat[(size_t)((b*LL + l)*NN + n)*DD + d];
            float wx = w * x;
            int e = shi[tok];
            if (e == 0) a0 += wx; else if (e == 1) a1 += wx; else a2 += wx;
        }
    }
    xbarA[(size_t)(0*16 + b)*DD + d] = (__bf16)a0;
    xbarA[(size_t)(1*16 + b)*DD + d] = (__bf16)a1;
    xbarA[(size_t)(2*16 + b)*DD + d] = (__bf16)a2;
}

// A-fragment loader: one contiguous 32B load per lane (k = 16*hi + j, matches packed B).
__device__ __forceinline__ v16bf load_afrag(const __bf16* A, int laneM, int hi, int k0) {
    return *(const v16bf*)(A + (size_t)laneM*DD + k0 + hi*16);
}

// ------------------------------------------------------------------
// Kernel 4: ybar[16,1024] = sum_e A_e @ We[e] + gatesum-weighted bias.
// One wave per 16-col tile; B from packed fragment layout (one 32B
// coalesced load per lane per k-step).
// ------------------------------------------------------------------
__global__ void k_expert_gemm(const __bf16* __restrict__ xbarA,  // [E][16][D]
                              const __bf16* __restrict__ WeP,    // packed, per e
                              const float*  __restrict__ be,
                              const float*  __restrict__ gsum,
                              __bf16* __restrict__ ybarA)        // [16][D]
{
    int lane  = threadIdx.x;
    int laneM = lane & 15;
    int hi    = lane >> 4;
    int tile  = blockIdx.x;
    v8f acc = {};
    for (int e = 0; e < EE; e++) {
        const __bf16* A  = xbarA + (size_t)e * 16 * DD;
        const __bf16* Bp = WeP   + (size_t)e * DD * DD
                         + ((size_t)tile * KB * 32 + lane) * 16;
        #pragma unroll 4
        for (int kb = 0; kb < KB; kb++) {
            v16bf a = load_afrag(A, laneM, hi, kb*32);
            v16bf b = *(const v16bf*)(Bp + (size_t)kb * 32 * 16);
            acc = __builtin_amdgcn_wmma_f32_16x16x32_bf16(
                      false, a, false, b, (short)0, acc, false, false);
        }
    }
    if (!hi) {
        #pragma unroll
        for (int r = 0; r < BB; r++) {
            int col = tile*16 + laneM;
            float bias = 0.f;
            for (int e = 0; e < EE; e++) bias += gsum[r*EE + e] * be[e*DD + col];
            ybarA[(size_t)r*DD + col] = (__bf16)(acc[r] + bias);
        }
    }
}

// ------------------------------------------------------------------
// Kernel 5: pooled[b,h] = (ybar[b] @ Wp + L*N*bp[h]) / cnt_b.
// ------------------------------------------------------------------
__global__ void k_proj(const __bf16* __restrict__ ybarA,   // [16][D]
                       const __bf16* __restrict__ WpP,     // packed
                       const float*  __restrict__ bp,
                       const float*  __restrict__ cntb,
                       float* __restrict__ pooled)
{
    int lane  = threadIdx.x;
    int laneM = lane & 15;
    int hi    = lane >> 4;
    int tile  = blockIdx.x;
    const __bf16* Bp = WpP + ((size_t)tile * KB * 32 + lane) * 16;
    v8f acc = {};
    #pragma unroll 4
    for (int kb = 0; kb < KB; kb++) {
        v16bf a = load_afrag(ybarA, laneM, hi, kb*32);
        v16bf b = *(const v16bf*)(Bp + (size_t)kb * 32 * 16);
        acc = __builtin_amdgcn_wmma_f32_16x16x32_bf16(
                  false, a, false, b, (short)0, acc, false, false);
    }
    if (!hi) {
        #pragma unroll
        for (int r = 0; r < BB; r++) {
            int col = tile*16 + laneM;
            pooled[(size_t)r*HH + col] = (acc[r] + (float)TPB * bp[col]) / cntb[r];
        }
    }
}

// ------------------------------------------------------------------
extern "C" void kernel_launch(void* const* d_in, const int* in_sizes, int n_in,
                              void* d_out, int out_size, void* d_ws, size_t ws_size,
                              hipStream_t stream) {
    (void)in_sizes; (void)n_in; (void)out_size; (void)ws_size;
    const float*         feat = (const float*)d_in[0];
    const unsigned char* mask = (const unsigned char*)d_in[1];
    const float*         wg   = (const float*)d_in[2];
    const float*         We   = (const float*)d_in[3];
    const float*         be   = (const float*)d_in[4];
    const float*         Wp   = (const float*)d_in[5];
    const float*         bp   = (const float*)d_in[6];
    float* out = (float*)d_out;             // pooled [4*4096] then l_aux [1]

    char* ws = (char*)d_ws;
    auto alloc = [&](size_t bytes) {
        char* p = ws;
        ws += (bytes + 255) & ~(size_t)255;
        return p;
    };
    float*  gates    = (float*) alloc((size_t)SS * 3 * sizeof(float));
    float*  gate_top = (float*) alloc((size_t)SS * sizeof(float));
    int*    eidx     = (int*)   alloc((size_t)SS * sizeof(int));
    float*  wgate    = (float*) alloc((size_t)SS * sizeof(float));
    float*  gsum     = (float*) alloc((size_t)BB * EE * sizeof(float));
    float*  cntb     = (float*) alloc((size_t)BB * sizeof(float));
    __bf16* xbarA    = (__bf16*)alloc((size_t)EE * 16 * DD * sizeof(__bf16));
    __bf16* ybarA    = (__bf16*)alloc((size_t)16 * DD * sizeof(__bf16));
    __bf16* WeP      = (__bf16*)alloc((size_t)EE * DD * DD * sizeof(__bf16));
    __bf16* WpP      = (__bf16*)alloc((size_t)DD * HH * sizeof(__bf16));

    // zero padded A matrices (rows 4..15 stay zero)
    int nx = (int)((size_t)EE*16*DD*sizeof(__bf16) / 4);
    int ny = (int)((size_t)16*DD*sizeof(__bf16) / 4);
    k_zero<<<(nx + 255)/256, 256, 0, stream>>>((unsigned int*)xbarA, nx);
    k_zero<<<(ny + 255)/256, 256, 0, stream>>>((unsigned int*)ybarA, ny);

    // pack weights to bf16 fragment layout
    for (int e = 0; e < EE; e++)
        k_pack<<<(DD/16)*KB, 256, 0, stream>>>(We + (size_t)e*DD*DD,
                                               WeP + (size_t)e*DD*DD, DD, KB);
    k_pack<<<(HH/16)*KB, 256, 0, stream>>>(Wp, WpP, HH, KB);

    k_gating<<<SS/8, 256, 0, stream>>>(feat, wg, gates, gate_top, eidx);
    k_scan  <<<1,    256, 0, stream>>>(gates, gate_top, eidx, mask,
                                       wgate, gsum, cntb, out + BB*HH);
    k_xbar  <<<BB*4, 256, 0, stream>>>(feat, wgate, eidx, xbarA);
    k_expert_gemm<<<DD/16, 32, 0, stream>>>(xbarA, WeP, be, gsum, ybarA);
    k_proj       <<<HH/16, 32, 0, stream>>>(ybarA, WpP, bp, cntb, out);
}
